// ComPoM_75118978007438
// MI455X (gfx1250) — matrix operand order repaired and checked by reading, not compiled, not run on hardware
//
#include <hip/hip_runtime.h>

// Problem constants (match reference)
#define B_   4
#define T_   4096
#define N_   4096
#define DIM_ 1024
#define DE_  2048

typedef _Float16 v16h __attribute__((ext_vector_type(16)));
typedef _Float16 v8h  __attribute__((ext_vector_type(8)));
typedef float    v8f  __attribute__((ext_vector_type(8)));

__device__ __forceinline__ v8f vzero8() {
    v8f z = {0.f, 0.f, 0.f, 0.f, 0.f, 0.f, 0.f, 0.f};
    return z;
}

// ---------------------------------------------------------------------------
// A-fragment (16x32, f16) from row-major fp32 source, converting inline.
// ISA layout: lanes 0-15 hold row M=lane, K {k0..k0+7, k0+16..k0+23};
//             lanes 16-31 hold row M=lane-16, K {k0+8..k0+15, k0+24..k0+31}.
// ---------------------------------------------------------------------------
__device__ __forceinline__ v16h load_a_f32(const float* __restrict__ base,
                                           int ld, int m0, int k0, int lane) {
    const int row = m0 + (lane & 15);
    const int ks  = k0 + ((lane & 16) ? 8 : 0);
    const float4* p = reinterpret_cast<const float4*>(base + (size_t)row * ld + ks);
    float4 f0 = p[0];
    float4 f1 = p[1];
    const float4* q = reinterpret_cast<const float4*>(base + (size_t)row * ld + ks + 16);
    float4 f2 = q[0];
    float4 f3 = q[1];
    v16h a;
    a[0]  = (_Float16)f0.x; a[1]  = (_Float16)f0.y; a[2]  = (_Float16)f0.z; a[3]  = (_Float16)f0.w;
    a[4]  = (_Float16)f1.x; a[5]  = (_Float16)f1.y; a[6]  = (_Float16)f1.z; a[7]  = (_Float16)f1.w;
    a[8]  = (_Float16)f2.x; a[9]  = (_Float16)f2.y; a[10] = (_Float16)f2.z; a[11] = (_Float16)f2.w;
    a[12] = (_Float16)f3.x; a[13] = (_Float16)f3.y; a[14] = (_Float16)f3.z; a[15] = (_Float16)f3.w;
    return a;
}

// A-fragment (16x32, f16) from row-major f16 source (two b128 loads).
__device__ __forceinline__ v16h load_a_f16(const _Float16* __restrict__ base,
                                           int ld, int m0, int k0, int lane) {
    const int row = m0 + (lane & 15);
    const int ks  = k0 + ((lane & 16) ? 8 : 0);
    const v8h* p = reinterpret_cast<const v8h*>(base + (size_t)row * ld + ks);
    v8h lo = p[0];
    const v8h* q = reinterpret_cast<const v8h*>(base + (size_t)row * ld + ks + 16);
    v8h hi = q[0];
    v16h a;
#pragma unroll
    for (int i = 0; i < 8; ++i) { a[i] = lo[i]; a[8 + i] = hi[i]; }
    return a;
}

// ---------------------------------------------------------------------------
// B-fragment (32x16, f16) where B[k,n] = W[e0+n, k], W row-major (E x K) f16.
// ISA layout: lanes 0-15 hold col n=lane, K {k0..k0+7, k0+16..k0+23};
//             lanes 16-31 hold col n=lane-16, K {k0+8..k0+15, k0+24..k0+31}.
// -> each lane: two 8-f16 runs from its weight row = two b128 loads.
// ---------------------------------------------------------------------------
__device__ __forceinline__ v16h load_b_f16(const _Float16* __restrict__ W,
                                           int ldk, int e0, int k0, int lane) {
    const int e  = e0 + (lane & 15);
    const int ks = k0 + ((lane & 16) ? 8 : 0);
    const v8h* p = reinterpret_cast<const v8h*>(W + (size_t)e * ldk + ks);
    v8h lo = p[0];
    v8h hi = p[2];
    v16h b;
#pragma unroll
    for (int i = 0; i < 8; ++i) { b[i] = lo[i]; b[8 + i] = hi[i]; }
    return b;
}

// ---------------------------------------------------------------------------
// Kernel: zero scratch
// ---------------------------------------------------------------------------
__global__ void k_zero(float* __restrict__ p, int n) {
    int i = blockIdx.x * blockDim.x + threadIdx.x;
    if (i < n) p[i] = 0.f;
}

// ---------------------------------------------------------------------------
// Kernel: fp32 -> f16 conversion (8 elements / thread), for the weights.
// ---------------------------------------------------------------------------
__global__ void k_cvt(const float* __restrict__ src, _Float16* __restrict__ dst,
                      int n8) {
    int i = blockIdx.x * blockDim.x + threadIdx.x;
    if (i < n8) {
        const float4* s = reinterpret_cast<const float4*>(src) + 2 * (size_t)i;
        float4 f0 = s[0];
        float4 f1 = s[1];
        v8h o;
        o[0] = (_Float16)f0.x; o[1] = (_Float16)f0.y; o[2] = (_Float16)f0.z; o[3] = (_Float16)f0.w;
        o[4] = (_Float16)f1.x; o[5] = (_Float16)f1.y; o[6] = (_Float16)f1.z; o[7] = (_Float16)f1.w;
        reinterpret_cast<v8h*>(dst)[i] = o;
    }
}

// ---------------------------------------------------------------------------
// Kernel 1: hmsum[b,e] += sum_n mask[b,n] * poly(act( (xc @ Wpo^T)[b,n,e] ))
// Wave tile: 32 n-rows x 64 e-cols (2 A-frags x 4 B-frags = 8 WMMA / K-step),
// 4 tiles (128 rows) per wave; register column sums -> shfl pair-reduce ->
// one float atomicAdd per (b,e) partial.
// ---------------------------------------------------------------------------
__global__ __launch_bounds__(128) void k_hm(const float* __restrict__ xc,
                                            const int* __restrict__ mask,
                                            const _Float16* __restrict__ Wpo16,
                                            const float* __restrict__ coeff,
                                            float* __restrict__ hmsum) {
    const int lane = threadIdx.x & 31;
    const int wave = threadIdx.x >> 5;
    const int b    = blockIdx.z;
    const int e0   = (blockIdx.x * 4 + wave) * 64;
    const int n0   = blockIdx.y * 128;

    const float* xcb   = xc + (size_t)b * N_ * DIM_;
    const int*   maskb = mask + (size_t)b * N_;

    float c0[4], c1[4], c2[4];
#pragma unroll
    for (int j = 0; j < 4; ++j) {
        int e = e0 + 16 * j + (lane & 15);
        c0[j] = coeff[e * 3 + 0];
        c1[j] = coeff[e * 3 + 1];
        c2[j] = coeff[e * 3 + 2];
    }

    float colsum[4] = {0.f, 0.f, 0.f, 0.f};
    const int mhalf = (lane & 16) ? 8 : 0;

    for (int nt = 0; nt < 4; ++nt) {
        const int m0 = n0 + nt * 32;
        v8f acc0[4], acc1[4];
#pragma unroll
        for (int j = 0; j < 4; ++j) { acc0[j] = vzero8(); acc1[j] = vzero8(); }

        for (int k0 = 0; k0 < DIM_; k0 += 32) {
            v16h a0 = load_a_f32(xcb, DIM_, m0,      k0, lane);
            v16h a1 = load_a_f32(xcb, DIM_, m0 + 16, k0, lane);
#pragma unroll
            for (int j = 0; j < 4; ++j) {
                v16h bm = load_b_f16(Wpo16, DIM_, e0 + 16 * j, k0, lane);
                acc0[j] = __builtin_amdgcn_wmma_f32_16x16x32_f16(
                    false, a0, false, bm, (short)0, acc0[j], false, false);
                acc1[j] = __builtin_amdgcn_wmma_f32_16x16x32_f16(
                    false, a1, false, bm, (short)0, acc1[j], false, false);
            }
        }

        // mask weights for this lane's rows in both 16-row groups
        const int4* mp0 = reinterpret_cast<const int4*>(maskb + m0 + mhalf);
        const int4* mp1 = reinterpret_cast<const int4*>(maskb + m0 + 16 + mhalf);
        int4 mA0 = mp0[0], mB0 = mp0[1];
        int4 mA1 = mp1[0], mB1 = mp1[1];
        float mw0[8], mw1[8];
        mw0[0] = (float)mA0.x; mw0[1] = (float)mA0.y; mw0[2] = (float)mA0.z; mw0[3] = (float)mA0.w;
        mw0[4] = (float)mB0.x; mw0[5] = (float)mB0.y; mw0[6] = (float)mB0.z; mw0[7] = (float)mB0.w;
        mw1[0] = (float)mA1.x; mw1[1] = (float)mA1.y; mw1[2] = (float)mA1.z; mw1[3] = (float)mA1.w;
        mw1[4] = (float)mB1.x; mw1[5] = (float)mB1.y; mw1[6] = (float)mB1.z; mw1[7] = (float)mB1.w;

#pragma unroll
        for (int j = 0; j < 4; ++j) {
#pragma unroll
            for (int r = 0; r < 8; ++r) {
                float h0 = acc0[j][r];
                float a0 = (h0 >= 0.f) ? h0 : 0.01f * h0;      // leaky_relu
                a0 = fminf(fmaxf(a0, -0.1f), 6.0f);            // clamp
                colsum[j] += mw0[r] * (a0 * (c0[j] + a0 * (c1[j] + a0 * c2[j])));

                float h1 = acc1[j][r];
                float a1 = (h1 >= 0.f) ? h1 : 0.01f * h1;
                a1 = fminf(fmaxf(a1, -0.1f), 6.0f);
                colsum[j] += mw1[r] * (a1 * (c0[j] + a1 * (c1[j] + a1 * c2[j])));
            }
        }
    }

    // lanes l and l+16 hold the same column -> pair-reduce, then atomic.
#pragma unroll
    for (int j = 0; j < 4; ++j) {
        float tot = colsum[j] + __shfl_xor(colsum[j], 16, 32);
        if (lane < 16)
            atomicAdd(hmsum + (size_t)b * DE_ + e0 + 16 * j + lane, tot);
    }
}

// ---------------------------------------------------------------------------
// Kernel 2: hm[b,e] = hmsum[b,e] / sum_n mask[b,n]
// ---------------------------------------------------------------------------
__global__ __launch_bounds__(256) void k_fin(const int* __restrict__ mask,
                                             const float* __restrict__ hmsum,
                                             float* __restrict__ hm) {
    __shared__ float scnt[256];
    const int b = blockIdx.x;
    const int t = threadIdx.x;
    int s = 0;
    for (int n = t; n < N_; n += 256) s += mask[(size_t)b * N_ + n];
    scnt[t] = (float)s;
    __syncthreads();
    for (int off = 128; off > 0; off >>= 1) {
        if (t < off) scnt[t] += scnt[t + off];
        __syncthreads();
    }
    const float inv = 1.0f / scnt[0];
    for (int e = t; e < DE_; e += 256)
        hm[(size_t)b * DE_ + e] = hmsum[(size_t)b * DE_ + e] * inv;
}

// ---------------------------------------------------------------------------
// Kernel 3: sh[b,t,e] = hardsigmoid((xq @ Wse^T)[b,t,e] + bse[e]) * hm[b,e]
// Wave tile: 32 t-rows x 64 e-cols, 2 tiles (64 rows) per wave. f16 output.
// ---------------------------------------------------------------------------
__global__ __launch_bounds__(128) void k_gate(const float* __restrict__ xq,
                                              const _Float16* __restrict__ Wse16,
                                              const float* __restrict__ bse,
                                              const float* __restrict__ hm,
                                              _Float16* __restrict__ sh) {
    const int lane = threadIdx.x & 31;
    const int wave = threadIdx.x >> 5;
    const int b    = blockIdx.z;
    const int e0   = (blockIdx.x * 4 + wave) * 64;
    const int t0   = blockIdx.y * 64;

    const float* xqb = xq + (size_t)b * T_ * DIM_;

    float bsev[4], hmv[4];
#pragma unroll
    for (int j = 0; j < 4; ++j) {
        int e = e0 + 16 * j + (lane & 15);
        bsev[j] = bse[e];
        hmv[j]  = hm[(size_t)b * DE_ + e];
    }

    const int mhalf = (lane & 16) ? 8 : 0;

    for (int mt = 0; mt < 2; ++mt) {
        const int m0 = t0 + mt * 32;
        v8f acc0[4], acc1[4];
#pragma unroll
        for (int j = 0; j < 4; ++j) { acc0[j] = vzero8(); acc1[j] = vzero8(); }

        for (int k0 = 0; k0 < DIM_; k0 += 32) {
            v16h a0 = load_a_f32(xqb, DIM_, m0,      k0, lane);
            v16h a1 = load_a_f32(xqb, DIM_, m0 + 16, k0, lane);
#pragma unroll
            for (int j = 0; j < 4; ++j) {
                v16h bm = load_b_f16(Wse16, DIM_, e0 + 16 * j, k0, lane);
                acc0[j] = __builtin_amdgcn_wmma_f32_16x16x32_f16(
                    false, a0, false, bm, (short)0, acc0[j], false, false);
                acc1[j] = __builtin_amdgcn_wmma_f32_16x16x32_f16(
                    false, a1, false, bm, (short)0, acc1[j], false, false);
            }
        }

#pragma unroll
        for (int j = 0; j < 4; ++j) {
            const int e = e0 + 16 * j + (lane & 15);
#pragma unroll
            for (int r = 0; r < 8; ++r) {
                const int m = m0 + r + mhalf;
                float v0  = acc0[j][r] + bsev[j];
                float sg0 = fminf(fmaxf(v0 * (1.0f / 6.0f) + 0.5f, 0.f), 1.f);
                sh[((size_t)b * T_ + m) * DE_ + e] = (_Float16)(sg0 * hmv[j]);

                float v1  = acc1[j][r] + bsev[j];
                float sg1 = fminf(fmaxf(v1 * (1.0f / 6.0f) + 0.5f, 0.f), 1.f);
                sh[((size_t)b * T_ + m + 16) * DE_ + e] = (_Float16)(sg1 * hmv[j]);
            }
        }
    }
}

// ---------------------------------------------------------------------------
// Kernel 4: out[b,t,d] = (sh @ Wag^T)[b,t,d]   (sh f16, Wag16 f16, fp32 acc)
// Wave tile: 32 t-rows x 64 d-cols, 2 tiles (64 rows) per wave.
// ---------------------------------------------------------------------------
__global__ __launch_bounds__(128) void k_out(const _Float16* __restrict__ sh,
                                             const _Float16* __restrict__ Wag16,
                                             float* __restrict__ out) {
    const int lane = threadIdx.x & 31;
    const int wave = threadIdx.x >> 5;
    const int b    = blockIdx.z;
    const int d0   = (blockIdx.x * 4 + wave) * 64;
    const int t0   = blockIdx.y * 64;

    const _Float16* shb = sh + (size_t)b * T_ * DE_;
    const int mhalf = (lane & 16) ? 8 : 0;

    for (int mt = 0; mt < 2; ++mt) {
        const int m0 = t0 + mt * 32;
        v8f acc0[4], acc1[4];
#pragma unroll
        for (int j = 0; j < 4; ++j) { acc0[j] = vzero8(); acc1[j] = vzero8(); }

        for (int k0 = 0; k0 < DE_; k0 += 32) {
            v16h a0 = load_a_f16(shb, DE_, m0,      k0, lane);
            v16h a1 = load_a_f16(shb, DE_, m0 + 16, k0, lane);
#pragma unroll
            for (int j = 0; j < 4; ++j) {
                v16h bm = load_b_f16(Wag16, DE_, d0 + 16 * j, k0, lane);
                acc0[j] = __builtin_amdgcn_wmma_f32_16x16x32_f16(
                    false, a0, false, bm, (short)0, acc0[j], false, false);
                acc1[j] = __builtin_amdgcn_wmma_f32_16x16x32_f16(
                    false, a1, false, bm, (short)0, acc1[j], false, false);
            }
        }

#pragma unroll
        for (int j = 0; j < 4; ++j) {
            const int d = d0 + 16 * j + (lane & 15);
#pragma unroll
            for (int r = 0; r < 8; ++r) {
                const int m = m0 + r + mhalf;
                out[((size_t)b * T_ + m) * DIM_ + d]        = acc0[j][r];
                out[((size_t)b * T_ + m + 16) * DIM_ + d]   = acc1[j][r];
            }
        }
    }
}

// ---------------------------------------------------------------------------
// Launcher
// inputs (setup_inputs order): xq, xc, mask, Wpo, Wse, bse, coeff, Wag
// ---------------------------------------------------------------------------
extern "C" void kernel_launch(void* const* d_in, const int* in_sizes, int n_in,
                              void* d_out, int out_size, void* d_ws, size_t ws_size,
                              hipStream_t stream) {
    const float* xq    = (const float*)d_in[0];
    const float* xc    = (const float*)d_in[1];
    const int*   mask  = (const int*)d_in[2];
    const float* Wpo   = (const float*)d_in[3];
    const float* Wse   = (const float*)d_in[4];
    const float* bse   = (const float*)d_in[5];
    const float* coeff = (const float*)d_in[6];
    const float* Wag   = (const float*)d_in[7];
    float* out = (float*)d_out;

    char* ws = (char*)d_ws;
    float*    hmsum = (float*)ws;                           // B*DE fp32 (32 KB)
    float*    hm    = (float*)(ws + (64u << 10));           // B*DE fp32 (32 KB)
    _Float16* Wpo16 = (_Float16*)(ws + (1u << 20));         // 4 MB
    _Float16* Wse16 = (_Float16*)(ws + (6u << 20));         // 4 MB
    _Float16* Wag16 = (_Float16*)(ws + (11u << 20));        // 4 MB
    _Float16* sh    = (_Float16*)(ws + (16u << 20));        // B*T*DE f16 (64 MiB)

    // 0) one-time f16 weight copies (tiny: 3 x 2M elements)
    const int nw8 = (DE_ * DIM_) / 8;   // same element count for all three
    k_cvt<<<dim3((nw8 + 255) / 256), 256, 0, stream>>>(Wpo, Wpo16, nw8);
    k_cvt<<<dim3((nw8 + 255) / 256), 256, 0, stream>>>(Wse, Wse16, nw8);
    k_cvt<<<dim3((nw8 + 255) / 256), 256, 0, stream>>>(Wag, Wag16, nw8);

    // 1) zero hm partial sums
    k_zero<<<dim3((B_ * DE_ + 255) / 256), 256, 0, stream>>>(hmsum, B_ * DE_);

    // 2) GEMM1 + activation + poly + masked column sums
    k_hm<<<dim3(DE_ / 256, N_ / 128, B_), 128, 0, stream>>>(xc, mask, Wpo16, coeff, hmsum);

    // 3) finalize masked mean
    k_fin<<<dim3(B_), 256, 0, stream>>>(mask, hmsum, hm);

    // 4) GEMM2 + hardsigmoid + gate-by-hm -> sh (f16)
    k_gate<<<dim3(DE_ / 256, T_ / 64, B_), 128, 0, stream>>>(xq, Wse16, bse, hm, sh);

    // 5) GEMM3 -> out (fp32)
    k_out<<<dim3(DIM_ / 256, T_ / 64, B_), 128, 0, stream>>>(sh, Wag16, out);
}
